// CrossAttFeatTrans_62672162783793
// MI455X (gfx1250) — matrix operand order repaired and checked by reading, not compiled, not run on hardware
//
#include <hip/hip_runtime.h>
#include <cstdint>
#include <cstddef>

#define B_   2
#define U1_  3072
#define U2_  3072
#define IF_  256
#define FD_  256
#define M_   4
#define DH_  64

typedef __bf16 v16bf __attribute__((ext_vector_type(16)));
typedef float  v8f   __attribute__((ext_vector_type(8)));

union AV {
  v16bf v;
  unsigned int u[8];
  unsigned short s[16];
};

__device__ __forceinline__ v8f wmma_bf16(v16bf a, v16bf b, v8f c) {
  return __builtin_amdgcn_wmma_f32_16x16x32_bf16(false, a, false, b, (short)0, c,
                                                 false, false);
}

__device__ __forceinline__ unsigned short f2bf(float f) {
  unsigned int x;
  __builtin_memcpy(&x, &f, 4);
  x += 0x7FFFu + ((x >> 16) & 1u);   // round-to-nearest-even
  return (unsigned short)(x >> 16);
}

// A-operand (16x32 bf16) maps: element j -> K = 16h + j ; dword dj -> 8h + dj
// in a K-contiguous row.
__device__ __forceinline__ int a_dmap(int h, int dj) {
  return h * 8 + dj;
}
// B-operand (32x16 bf16): element j -> K = 16h + j ; dword dj -> 8h + dj.

// CDNA5 async global -> LDS copy (ASYNCcnt-tracked), GV addressing mode.
__device__ __forceinline__ void async_load_b128(unsigned lds_byte_off,
                                                const void* gptr) {
  unsigned long long ga = (unsigned long long)(uintptr_t)gptr;
  asm volatile("global_load_async_to_lds_b128 %0, %1, off"
               :
               : "v"(lds_byte_off), "v"(ga)
               : "memory");
}
__device__ __forceinline__ void wait_asynccnt0() {
  asm volatile("s_wait_asynccnt 0x0" ::: "memory");
}

// ---------------------------------------------------------------------------
// 0) elementwise f32 -> bf16 convert
// ---------------------------------------------------------------------------
__global__ void f32_to_bf16_kernel(const float* __restrict__ in,
                                   unsigned short* __restrict__ out, int n) {
  int i = blockIdx.x * 256 + threadIdx.x;
  if (i < n) out[i] = f2bf(in[i]);
}

// ---------------------------------------------------------------------------
// 1) projection GEMM: out = X @ W^T (+bias).  X:[rows][256] bf16, W:[OF][256]
//    mode 0: q/k style write  -> out[((b*4+m)*U1+u)*64 + d],  of = m*64+d
//    mode 2: v transposed     -> out[((b*4+m)*256+f)*U2 + u], of = m*256+f
// ---------------------------------------------------------------------------
__global__ void __launch_bounds__(32) proj_kernel(
    const unsigned short* __restrict__ X, const unsigned short* __restrict__ W,
    const float* __restrict__ bias, unsigned short* __restrict__ out,
    int ofTiles, int mode) {
  int rt = blockIdx.x / ofTiles;
  int ot = blockIdx.x % ofTiles;
  int lane = threadIdx.x & 31;
  int n = lane & 15, h = lane >> 4;

  const unsigned int* xr = (const unsigned int*)X + (size_t)(rt * 16 + n) * 128;
  const unsigned int* wr = (const unsigned int*)W + (size_t)(ot * 16 + n) * 128;

  v8f c = {0.f, 0.f, 0.f, 0.f, 0.f, 0.f, 0.f, 0.f};
#pragma unroll
  for (int ic = 0; ic < 8; ++ic) {
    AV a, b;
#pragma unroll
    for (int dj = 0; dj < 8; ++dj) {
      a.u[dj] = xr[ic * 16 + a_dmap(h, dj)];
      b.u[dj] = wr[ic * 16 + h * 8 + dj];
    }
    c = wmma_bf16(a.v, b.v, c);
  }

  int of = ot * 16 + n;
  float bi = bias ? bias[of] : 0.0f;
#pragma unroll
  for (int r = 0; r < 8; ++r) {
    int rowg = rt * 16 + r + 8 * h;
    int bb = rowg / U1_;
    int u = rowg % U1_;
    float val = c[r] + bi;
    size_t idx;
    if (mode == 2) {
      int m = of >> 8, f = of & 255;
      idx = ((size_t)(bb * M_ + m) * FD_ + f) * (size_t)U2_ + u;
    } else {
      int m = of >> 6, d = of & 63;
      idx = ((size_t)(bb * M_ + m) * U1_ + u) * (size_t)DH_ + d;
    }
    out[idx] = f2bf(val);
  }
}

// ---------------------------------------------------------------------------
// 2) flash attention.  4 waves/block share (b,m); each wave owns a 16-row q
//    tile; 32-key chunks of k and v^T double-buffered in LDS via CDNA5 async
//    global->LDS copies; online softmax in f32.
// ---------------------------------------------------------------------------
#define NCH (U2_ / 32)   // 96 chunks

__global__ void __launch_bounds__(128) attn_kernel(
    const unsigned short* __restrict__ qb,   // [bm][u1][64] bf16
    const unsigned short* __restrict__ kbuf, // [bm][u2][64] bf16
    const unsigned short* __restrict__ vt,   // [bm][256][u2] bf16
    float* __restrict__ fused)               // [bm][u1][256] f32
{
  __shared__ unsigned int kC[2 * 1024];   // 2 x (32 keys x 64 d) bf16
  __shared__ unsigned int vC[2 * 4096];   // 2 x (256 f x 32 keys) bf16
  __shared__ float scW[4][16 * 33];
  __shared__ float alW[4][16];

  int bm = blockIdx.x / 48;
  int tg = blockIdx.x % 48;
  int w = threadIdx.x >> 5;
  int lane = threadIdx.x & 31;
  int n = lane & 15, h = lane >> 4;
  int tid = threadIdx.x;
  int u0 = (tg * 4 + w) * 16;

  unsigned kC_base = (unsigned)(uintptr_t)(&kC[0]);
  unsigned vC_base = (unsigned)(uintptr_t)(&vC[0]);
  const char* kg_all = (const char*)kbuf + (size_t)bm * U2_ * 128;   // 64 bf16/row
  const char* vg_all = (const char*)vt + (size_t)bm * FD_ * U2_ * 2; // rows of 3072 bf16

  // issue async staging of chunk `cc` into buffer `buf`
  auto issue_chunk = [&](int cc, int buf) {
    int kv0 = cc * 32;
    const char* kg = kg_all + (size_t)kv0 * 128;  // 32 rows x 128B, contiguous
    unsigned klds = kC_base + (unsigned)buf * 4096u;
#pragma unroll
    for (int j = 0; j < 2; ++j) {
      int i = tid + j * 128;                       // 256 x 16B blocks
      async_load_b128(klds + (unsigned)i * 16u, kg + (size_t)i * 16);
    }
    const char* vg = vg_all + (size_t)kv0 * 2;     // column offset within row
    unsigned vlds = vC_base + (unsigned)buf * 16384u;
#pragma unroll
    for (int j = 0; j < 8; ++j) {
      int i = tid + j * 128;                       // 1024 x 16B blocks
      int f = i >> 2, sb = i & 3;                  // row f, 16B sub-block
      async_load_b128(vlds + (unsigned)i * 16u,
                      vg + (size_t)f * (U2_ * 2) + (size_t)sb * 16);
    }
  };

  // q A-operands, loaded once (K = d, two 32-wide halves of d=64)
  const unsigned int* qu =
      (const unsigned int*)qb + ((size_t)bm * U1_ + u0 + n) * 32;
  AV qa0, qa1;
#pragma unroll
  for (int dj = 0; dj < 8; ++dj) {
    qa0.u[dj] = qu[a_dmap(h, dj)];
    qa1.u[dj] = qu[16 + a_dmap(h, dj)];
  }

  v8f zero = {0.f, 0.f, 0.f, 0.f, 0.f, 0.f, 0.f, 0.f};
  v8f acc[16];
#pragma unroll
  for (int fg = 0; fg < 16; ++fg) acc[fg] = zero;
  float mrun = -3.0e38f, lrun = 0.0f;

  issue_chunk(0, 0);  // prologue

  for (int ch = 0; ch < NCH; ++ch) {
    int buf = ch & 1;
    const unsigned int* kCb = kC + buf * 1024;
    const unsigned int* vCb = vC + buf * 4096;

    wait_asynccnt0();   // this wave's chunk-`ch` async copies landed
    __syncthreads();    // everyone's landed; everyone done reading buf^1
    if (ch + 1 < NCH) issue_chunk(ch + 1, buf ^ 1);  // overlap with compute

    // scores = q @ k^T (two 16-key groups, chained over d)
#pragma unroll
    for (int g = 0; g < 2; ++g) {
      AV b0, b1;
      int keyrow = g * 16 + n;
#pragma unroll
      for (int dj = 0; dj < 8; ++dj) {
        b0.u[dj] = kCb[keyrow * 32 + h * 8 + dj];
        b1.u[dj] = kCb[keyrow * 32 + 16 + h * 8 + dj];
      }
      v8f sc = wmma_bf16(qa0.v, b0.v, zero);
      sc = wmma_bf16(qa1.v, b1.v, sc);
#pragma unroll
      for (int r = 0; r < 8; ++r)
        scW[w][(r + 8 * h) * 33 + g * 16 + n] = sc[r] * 0.125f;  // 1/sqrt(64)
    }
    __syncthreads();

    // per-row online softmax (row = n, duplicated across lane halves)
    float s[32];
    float cmax = -3.0e38f;
#pragma unroll
    for (int k = 0; k < 32; ++k) {
      s[k] = scW[w][n * 33 + k];
      cmax = fmaxf(cmax, s[k]);
    }
    float mnew = fmaxf(mrun, cmax);
    float alpha = __expf(mrun - mnew);
    float csum = 0.0f;
#pragma unroll
    for (int k = 0; k < 32; ++k) {
      s[k] = __expf(s[k] - mnew);
      csum += s[k];
    }
    lrun = lrun * alpha + csum;
    mrun = mnew;
    if (lane < 16) alW[w][lane] = alpha;

    // pack probs directly into A-operand layout (lane = its own row)
    AV pa;
#pragma unroll
    for (int j = 0; j < 16; ++j) {
      int base = j;
      float pv = h ? s[base + 16] : s[base];
      pa.s[j] = f2bf(pv);
    }
    __syncthreads();

    float al[8];
#pragma unroll
    for (int r = 0; r < 8; ++r) al[r] = alW[w][r + 8 * h];
#pragma unroll
    for (int fg = 0; fg < 16; ++fg) {
      AV vb;
#pragma unroll
      for (int dj = 0; dj < 8; ++dj)
        vb.u[dj] = vCb[(fg * 16 + n) * 16 + h * 8 + dj];
      v8f a = acc[fg];
#pragma unroll
      for (int r = 0; r < 8; ++r) a[r] *= al[r];
      acc[fg] = wmma_bf16(pa.v, vb.v, a);
    }
  }

  float linv = 1.0f / lrun;
  __syncthreads();
  if (lane < 16) alW[w][lane] = linv;
  __syncthreads();
  float al[8];
#pragma unroll
  for (int r = 0; r < 8; ++r) al[r] = alW[w][r + 8 * h];
  float* fo = fused + ((size_t)bm * U1_ + u0) * FD_;
#pragma unroll
  for (int fg = 0; fg < 16; ++fg)
#pragma unroll
    for (int r = 0; r < 8; ++r)
      fo[(size_t)(r + 8 * h) * FD_ + fg * 16 + n] = acc[fg][r] * al[r];
}

// ---------------------------------------------------------------------------
// 3) FFN + residual + LayerNorm + mode-score, one 16-row tile per wave
// ---------------------------------------------------------------------------
__global__ void __launch_bounds__(32) ffn_kernel(
    const float* __restrict__ fused, const unsigned short* __restrict__ Wm,
    const float* __restrict__ bmid, const unsigned short* __restrict__ Wo,
    const float* __restrict__ bout, const float* __restrict__ lng,
    const float* __restrict__ lnb, const float* __restrict__ Wagg,
    const float* __restrict__ bagg, float* __restrict__ outp,
    float* __restrict__ mscore) {
  __shared__ float fT[16 * 256];
  __shared__ unsigned int mT[16 * 128];  // 16x256 bf16

  int bm = blockIdx.x / 192;
  int u0 = (blockIdx.x % 192) * 16;
  int lane = threadIdx.x;
  int n = lane & 15, h = lane >> 4;
  v8f zero = {0.f, 0.f, 0.f, 0.f, 0.f, 0.f, 0.f, 0.f};

  const float* fsrc = fused + ((size_t)bm * U1_ + u0) * FD_;
  for (int i = lane; i < 4096; i += 32) fT[i] = fsrc[i];
  __syncthreads();

  AV a1[8];
#pragma unroll
  for (int ic = 0; ic < 8; ++ic)
#pragma unroll
    for (int j = 0; j < 16; ++j) {
      int kk = h * 16 + j;
      a1[ic].s[j] = f2bf(fT[n * 256 + ic * 32 + kk]);
    }

  unsigned short* mTs = (unsigned short*)mT;
#pragma unroll 1
  for (int og = 0; og < 16; ++og) {
    const unsigned int* wr =
        (const unsigned int*)Wm + (size_t)(og * 16 + n) * 128;
    v8f c = zero;
#pragma unroll
    for (int ic = 0; ic < 8; ++ic) {
      AV b;
#pragma unroll
      for (int dj = 0; dj < 8; ++dj) b.u[dj] = wr[ic * 16 + h * 8 + dj];
      c = wmma_bf16(a1[ic].v, b.v, c);
    }
    float bi = bmid[og * 16 + n];
#pragma unroll
    for (int r = 0; r < 8; ++r) {
      float x = c[r] + bi;
      float g = 0.5f * x * (1.0f + erff(x * 0.70710678118f));  // exact gelu
      mTs[(r + 8 * h) * 256 + og * 16 + n] = f2bf(g);
    }
  }
  __syncthreads();

  AV a2[8];
#pragma unroll
  for (int ic = 0; ic < 8; ++ic)
#pragma unroll
    for (int dj = 0; dj < 8; ++dj)
      a2[ic].u[dj] = mT[n * 128 + ic * 16 + a_dmap(h, dj)];

#pragma unroll 1
  for (int og = 0; og < 16; ++og) {
    const unsigned int* wr =
        (const unsigned int*)Wo + (size_t)(og * 16 + n) * 128;
    v8f c = zero;
#pragma unroll
    for (int ic = 0; ic < 8; ++ic) {
      AV b;
#pragma unroll
      for (int dj = 0; dj < 8; ++dj) b.u[dj] = wr[ic * 16 + h * 8 + dj];
      c = wmma_bf16(a2[ic].v, b.v, c);
    }
#pragma unroll
    for (int r = 0; r < 8; ++r) {
      int u = r + 8 * h;
      int f = og * 16 + n;
      fT[u * 256 + f] = c[r] + bout[f] + fT[u * 256 + f];  // residual, in place
    }
  }
  __syncthreads();

  if (lane < 16) {
    int u = lane;
    const float* row = fT + u * 256;
    float mu = 0.f;
    for (int f = 0; f < 256; ++f) mu += row[f];
    mu *= (1.0f / 256.0f);
    float var = 0.f;
    for (int f = 0; f < 256; ++f) {
      float d = row[f] - mu;
      var += d * d;
    }
    var *= (1.0f / 256.0f);
    float rinv = rsqrtf(var + 1e-12f);
    float ms = 0.f;
    float* op = outp + ((size_t)bm * U1_ + u0 + u) * FD_;
    for (int f = 0; f < 256; ++f) {
      float o = (row[f] - mu) * rinv * lng[f] + lnb[f];
      op[f] = o;
      ms += o * Wagg[f];
    }
    mscore[(size_t)bm * U1_ + u0 + u] = ms + bagg[0];
  }
}

// ---------------------------------------------------------------------------
// 4) softmax over modes + weighted sum
// ---------------------------------------------------------------------------
__global__ void agg_kernel(const float* __restrict__ outp,
                           const float* __restrict__ mscore,
                           float* __restrict__ out) {
  int idx = blockIdx.x * 256 + threadIdx.x;  // < B*U1*FD
  int b = idx / (U1_ * FD_);
  int rem = idx % (U1_ * FD_);
  int u = rem >> 8, f = rem & 255;
  float sc[4];
  float mx = -3.0e38f;
#pragma unroll
  for (int m = 0; m < 4; ++m) {
    sc[m] = mscore[(size_t)(b * 4 + m) * U1_ + u];
    mx = fmaxf(mx, sc[m]);
  }
  float se = 0.f;
#pragma unroll
  for (int m = 0; m < 4; ++m) {
    sc[m] = __expf(sc[m] - mx);
    se += sc[m];
  }
  float inv = 1.0f / se;
  float acc = 0.f;
#pragma unroll
  for (int m = 0; m < 4; ++m)
    acc += outp[((size_t)(b * 4 + m) * U1_ + u) * FD_ + f] * sc[m];
  out[idx] = acc * inv;
}

// ---------------------------------------------------------------------------
extern "C" void kernel_launch(void* const* d_in, const int* in_sizes, int n_in,
                              void* d_out, int out_size, void* d_ws,
                              size_t ws_size, hipStream_t stream) {
  const float* query = (const float*)d_in[0];
  const float* keyf  = (const float*)d_in[1];
  const float* Wq    = (const float*)d_in[2];
  const float* Wv    = (const float*)d_in[3];
  const float* bv    = (const float*)d_in[4];
  const float* Wm    = (const float*)d_in[5];
  const float* bmid  = (const float*)d_in[6];
  const float* Wo    = (const float*)d_in[7];
  const float* bout  = (const float*)d_in[8];
  const float* lng   = (const float*)d_in[9];
  const float* lnb   = (const float*)d_in[10];
  const float* Wagg  = (const float*)d_in[11];
  const float* bagg  = (const float*)d_in[12];
  float* out = (float*)d_out;

  char* ws = (char*)d_ws;
  auto alloc = [&](size_t bytes) -> char* {
    char* p = ws;
    ws += (bytes + 255) & ~(size_t)255;
    return p;
  };

  const size_t nXQ = (size_t)B_ * U1_ * IF_;
  const size_t nXK = (size_t)B_ * U2_ * IF_;
  unsigned short* xq_bf = (unsigned short*)alloc(nXQ * 2);
  unsigned short* xk_bf = (unsigned short*)alloc(nXK * 2);
  unsigned short* Wq_bf = (unsigned short*)alloc(256 * 256 * 2);
  unsigned short* Wv_bf = (unsigned short*)alloc(1024 * 256 * 2);
  unsigned short* Wm_bf = (unsigned short*)alloc(256 * 256 * 2);
  unsigned short* Wo_bf = (unsigned short*)alloc(256 * 256 * 2);
  unsigned short* q_bf  = (unsigned short*)alloc((size_t)B_ * M_ * U1_ * DH_ * 2);
  unsigned short* k_bf  = (unsigned short*)alloc((size_t)B_ * M_ * U2_ * DH_ * 2);
  unsigned short* v_t   = (unsigned short*)alloc((size_t)B_ * M_ * FD_ * U2_ * 2);
  float* fusedW  = (float*)alloc((size_t)B_ * M_ * U1_ * FD_ * 4);
  float* outpW   = (float*)alloc((size_t)B_ * M_ * U1_ * FD_ * 4);
  float* mscoreW = (float*)alloc((size_t)B_ * M_ * U1_ * 4);

  // 0) converts
  f32_to_bf16_kernel<<<(int)((nXQ + 255) / 256), 256, 0, stream>>>(query, xq_bf, (int)nXQ);
  f32_to_bf16_kernel<<<(int)((nXK + 255) / 256), 256, 0, stream>>>(keyf, xk_bf, (int)nXK);
  f32_to_bf16_kernel<<<(65536 + 255) / 256, 256, 0, stream>>>(Wq, Wq_bf, 65536);
  f32_to_bf16_kernel<<<(262144 + 255) / 256, 256, 0, stream>>>(Wv, Wv_bf, 262144);
  f32_to_bf16_kernel<<<(65536 + 255) / 256, 256, 0, stream>>>(Wm, Wm_bf, 65536);
  f32_to_bf16_kernel<<<(65536 + 255) / 256, 256, 0, stream>>>(Wo, Wo_bf, 65536);

  // 1) projections (tied QK weight)
  const int rowTiles = (B_ * U1_) / 16;  // 384
  proj_kernel<<<rowTiles * 16, 32, 0, stream>>>(xq_bf, Wq_bf, nullptr, q_bf, 16, 0);
  proj_kernel<<<rowTiles * 16, 32, 0, stream>>>(xk_bf, Wq_bf, nullptr, k_bf, 16, 0);
  proj_kernel<<<rowTiles * 64, 32, 0, stream>>>(xk_bf, Wv_bf, bv, v_t, 64, 2);

  // 2) flash attention -> fused (async double-buffered k/v staging)
  attn_kernel<<<B_ * M_ * 48, 128, 0, stream>>>(q_bf, k_bf, v_t, fusedW);

  // 3) FFN + LN + mode scores
  ffn_kernel<<<B_ * M_ * 192, 32, 0, stream>>>(fusedW, Wm_bf, bmid, Wo_bf, bout,
                                               lng, lnb, Wagg, bagg, outpW,
                                               mscoreW);

  // 4) aggregate over modes
  agg_kernel<<<(B_ * U1_ * FD_) / 256, 256, 0, stream>>>(outpW, mscoreW, out);
}
